// CFConv_84945863180598
// MI455X (gfx1250) — matrix operand order repaired and checked by reading, not compiled, not run on hardware
//
#include <hip/hip_runtime.h>

typedef __attribute__((ext_vector_type(16))) _Float16 v16h;
typedef __attribute__((ext_vector_type(8)))  _Float16 v8h;
typedef __attribute__((ext_vector_type(8)))  float    v8f;

#define B_    32
#define NA    128
#define NN    64
#define NG    64
#define NF    128
// padded LDS strides (elements)
#define SA_STR 72    // f16, 144B: 16B-aligned rows, bank shift 36
#define SH_STR 136   // f16, 272B: 16B-aligned rows, bank shift 4
#define SY_STR 132   // f32

__device__ __forceinline__ float sspf(float x) {
    // shifted softplus: log(1+exp(x)) - log(2), stable; hw transcendentals
    float r = fmaxf(x, 0.0f) + __logf(1.0f + __expf(-fabsf(x)));
    return r - 0.6931471805599453f;
}

__device__ __forceinline__ v8f wmma_f16(v16h a, v16h b, v8f c) {
    return __builtin_amdgcn_wmma_f32_16x16x32_f16(false, a, false, b, (short)0, c, false, false);
}

// Async 16B copy global -> LDS (ASYNCcnt), bypasses VGPRs.
__device__ __forceinline__ void async_copy16(void* lds_dst, const void* gsrc) {
    unsigned dst = (unsigned)(uintptr_t)lds_dst;   // LDS aperture: addr[31:0] = LDS offset
    asm volatile("global_load_async_to_lds_b128 %0, %1, off"
                 :: "v"(dst), "v"(gsrc) : "memory");
}
__device__ __forceinline__ void async_wait0() {
    asm volatile("s_wait_asynccnt 0x0" ::: "memory");
}

// A fragment (16x32 f16) from row-major f16 (LDS), rows m0..m0+15, K k0..k0+31.
// Lanes 0-15: row=lane, K 0..15; lanes 16-31: row=lane-16, K 16..31.
__device__ __forceinline__ v16h load_a(const _Float16* base, int stride, int m0, int k0, int lane) {
    int row = m0 + (lane & 15);
    int kb  = k0 + ((lane & 16) ? 16 : 0);
    const _Float16* p = base + row * stride + kb;
    v8h lo = *(const v8h*)(p);
    v8h hi = *(const v8h*)(p + 8);
    return __builtin_shufflevector(lo, hi, 0,1,2,3,4,5,6,7,8,9,10,11,12,13,14,15);
}

// B fragment (32x16 f16) from N-major transposed weights WT[n][k] (global).
// Lanes 0-15: col=lane K 0..15; lanes 16-31: col=lane-16 K 16..31 -> one 32B load.
__device__ __forceinline__ v16h load_b(const _Float16* WT, int strideK, int n0, int k0, int lane) {
    int col = n0 + (lane & 15);
    int kb  = k0 + ((lane & 16) ? 16 : 0);
    return *(const v16h*)(WT + col * strideK + kb);
}

// ---------------- prep: f16 casts + weight transposes ----------------
__global__ __launch_bounds__(256) void cfconv_prep(
    const float* __restrict__ x, const float* __restrict__ Win2f,
    const float* __restrict__ Wf1, const float* __restrict__ Wf2,
    const float* __restrict__ Wout,
    _Float16* __restrict__ xh, _Float16* __restrict__ Win2fT,
    _Float16* __restrict__ Wf1T, _Float16* __restrict__ Wf2T,
    float* __restrict__ WoutT)
{
    int i = blockIdx.x * 256 + threadIdx.x;
    if (i < B_ * NA * NF) xh[i] = (_Float16)x[i];
    if (i < NF * NF) {
        int n = i >> 7, k = i & 127;
        Win2fT[n * NF + k] = (_Float16)Win2f[k * NF + n];
        Wf2T[n * NF + k]   = (_Float16)Wf2[k * NF + n];
        WoutT[n * NF + k]  = Wout[k * NF + n];
    }
    if (i < NF * NG) {
        int n = i >> 6, k = i & 63;
        Wf1T[n * NG + k] = (_Float16)Wf1[k * NF + n];
    }
}

// ---------------- y = x @ W_in2f : (4096,128)x(128,128) WMMA GEMM ----------------
__global__ __launch_bounds__(256) void cfconv_ygemm(
    const _Float16* __restrict__ xh, const _Float16* __restrict__ Win2fT,
    float* __restrict__ y)
{
    __shared__ __align__(16) _Float16 sX[64 * SH_STR];
    int tid  = threadIdx.x;
    int lane = tid & 31, wave = tid >> 5;
    int row0wg = blockIdx.x * 64;

    // stage 64x128 f16 x-tile via async LDS copies (pure copy, no conversion)
    for (int i = tid; i < 64 * (NF / 8); i += 256) {
        int r = i >> 4, c8 = (i & 15) * 8;
        async_copy16(sX + r * SH_STR + c8, xh + (size_t)(row0wg + r) * NF + c8);
    }
    async_wait0();
    __syncthreads();

    int mt  = wave & 3;          // M tile 0..3
    int ng0 = (wave >> 2) * 4;   // N tile group 0 or 4
    v8f acc[4] = {};
    for (int k0 = 0; k0 < NF; k0 += 32) {
        v16h a = load_a(sX, SH_STR, mt * 16, k0, lane);
#pragma unroll
        for (int t = 0; t < 4; ++t) {
            v16h b = load_b(Win2fT, NF, (ng0 + t) * 16, k0, lane);
            acc[t] = wmma_f16(a, b, acc[t]);
        }
    }
    int mrow = row0wg + mt * 16 + ((lane & 16) ? 8 : 0);
#pragma unroll
    for (int t = 0; t < 4; ++t) {
        int N = (ng0 + t) * 16 + (lane & 15);
#pragma unroll
        for (int j = 0; j < 8; ++j)
            y[(size_t)(mrow + j) * NF + N] = acc[t][j];
    }
}

// ---------------- fused filter-net + cutoff + gather + aggregate + out ----------------
__global__ __launch_bounds__(256) void cfconv_main(
    const float* __restrict__ r_ij, const int* __restrict__ neighbors,
    const float* __restrict__ pmask, const float* __restrict__ f_ij,
    const float* __restrict__ b_f1, const float* __restrict__ b_f2,
    const float* __restrict__ b_out,
    const _Float16* __restrict__ Wf1T, const _Float16* __restrict__ Wf2T,
    const float* __restrict__ WoutT, const float* __restrict__ y,
    float* __restrict__ out)
{
    __shared__ __align__(16) _Float16 sA[NN * SA_STR];   // f_ij tile (f16)
    __shared__ __align__(16) _Float16 sH[NN * SH_STR];   // ssp(GEMM1) (f16)
    __shared__ __align__(16) float    sY[NN * SY_STR];   // gathered y rows
    __shared__ __align__(16) float    sAgg[NF];
    __shared__ float                  sCoef[NN];

    int pair = blockIdx.x;           // b*Na + a
    int b    = pair >> 7;            // NA == 128
    int tid  = threadIdx.x;
    int lane = tid & 31, wave = tid >> 5;

    // async gather of 64 neighbor rows of y into LDS (overlaps with work below)
    for (int i = tid; i < NN * (NF / 4); i += 256) {
        int n = i >> 5, c4 = (i & 31) * 4;
        int nbr = neighbors[(size_t)pair * NN + n];
        async_copy16(sY + n * SY_STR + c4, y + ((size_t)b * NA + nbr) * NF + c4);
    }

    if (tid < NF) sAgg[tid] = 0.0f;
    if (tid < NN) {
        float r = r_ij[(size_t)pair * NN + tid];
        float c = (r < 5.0f) ? 0.5f * (__cosf(r * 0.6283185307179586f) + 1.0f) : 0.0f;
        sCoef[tid] = c * pmask[(size_t)pair * NN + tid];
    }
    // f_ij tile: 64x64 f32 contiguous -> f16 LDS (needs conversion, VALU path)
    {
        const float4* src = (const float4*)(f_ij + (size_t)pair * NN * NG);
        for (int i = tid; i < NN * NG / 4; i += 256) {
            float4 v = src[i];
            int r = i >> 4, c = (i & 15) * 4;
            _Float16* d = sA + r * SA_STR + c;
            d[0] = (_Float16)v.x; d[1] = (_Float16)v.y;
            d[2] = (_Float16)v.z; d[3] = (_Float16)v.w;
        }
    }
    async_wait0();
    __syncthreads();

    int mt  = wave & 3;
    int ng0 = (wave >> 2) * 4;

    // GEMM1: (64x64 f_ij) x (64x128 W_f1) ; +b_f1 ; ssp -> sH (f16)
    {
        v8f acc[4] = {};
        for (int k0 = 0; k0 < NG; k0 += 32) {
            v16h a = load_a(sA, SA_STR, mt * 16, k0, lane);
#pragma unroll
            for (int t = 0; t < 4; ++t) {
                v16h bfrag = load_b(Wf1T, NG, (ng0 + t) * 16, k0, lane);
                acc[t] = wmma_f16(a, bfrag, acc[t]);
            }
        }
        int mrow = mt * 16 + ((lane & 16) ? 8 : 0);
#pragma unroll
        for (int t = 0; t < 4; ++t) {
            int N = (ng0 + t) * 16 + (lane & 15);
            float bb = b_f1[N];
#pragma unroll
            for (int j = 0; j < 8; ++j)
                sH[(mrow + j) * SH_STR + N] = (_Float16)sspf(acc[t][j] + bb);
        }
    }
    __syncthreads();

    // GEMM2: (64x128 sH) x (128x128 W_f2) ; +b_f2 ; *coef[n] ; *y_nb ; reduce over n
    {
        v8f acc[4] = {};
        for (int k0 = 0; k0 < NF; k0 += 32) {
            v16h a = load_a(sH, SH_STR, mt * 16, k0, lane);
#pragma unroll
            for (int t = 0; t < 4; ++t) {
                v16h bfrag = load_b(Wf2T, NF, (ng0 + t) * 16, k0, lane);
                acc[t] = wmma_f16(a, bfrag, acc[t]);
            }
        }
        int mrow = mt * 16 + ((lane & 16) ? 8 : 0);
#pragma unroll
        for (int t = 0; t < 4; ++t) {
            int N = (ng0 + t) * 16 + (lane & 15);
            float bb = b_f2[N];
            float s = 0.0f;
#pragma unroll
            for (int j = 0; j < 8; ++j) {
                int n = mrow + j;
                s += (acc[t][j] + bb) * sCoef[n] * sY[n * SY_STR + N];
            }
            atomicAdd(&sAgg[N], s);   // ds_add_f32
        }
    }
    __syncthreads();

    // out[o] = ssp(dot(agg, W_out[:,o]) + b_out[o]); 128x128 MACs on VALU
    if (tid < NF) {
        const float4* w4 = (const float4*)(WoutT + (size_t)tid * NF);
        const float4* a4 = (const float4*)sAgg;
        float s = b_out[tid];
#pragma unroll 8
        for (int f = 0; f < NF / 4; ++f) {
            float4 w = w4[f], a = a4[f];
            s += a.x * w.x + a.y * w.y + a.z * w.z + a.w * w.w;
        }
        out[(size_t)pair * NF + tid] = sspf(s);
    }
}

extern "C" void kernel_launch(void* const* d_in, const int* in_sizes, int n_in,
                              void* d_out, int out_size, void* d_ws, size_t ws_size,
                              hipStream_t stream) {
    const float* x        = (const float*)d_in[0];
    const float* r_ij     = (const float*)d_in[1];
    const int*   neighbors= (const int*)d_in[2];
    const float* pmask    = (const float*)d_in[3];
    const float* f_ij     = (const float*)d_in[4];
    const float* W_in2f   = (const float*)d_in[5];
    const float* W_f1     = (const float*)d_in[6];
    const float* b_f1     = (const float*)d_in[7];
    const float* W_f2     = (const float*)d_in[8];
    const float* b_f2     = (const float*)d_in[9];
    const float* W_out    = (const float*)d_in[10];
    const float* b_out    = (const float*)d_in[11];
    float* out = (float*)d_out;

    char* ws = (char*)d_ws;
    _Float16* xh     = (_Float16*)(ws);             // 4096*128*2   = 1048576
    _Float16* Win2fT = (_Float16*)(ws + 1048576);   // 128*128*2    = 32768
    _Float16* Wf1T   = (_Float16*)(ws + 1081344);   // 128*64*2     = 16384
    _Float16* Wf2T   = (_Float16*)(ws + 1097728);   // 128*128*2    = 32768
    float*    WoutT  = (float*)   (ws + 1130496);   // 128*128*4    = 65536
    float*    y      = (float*)   (ws + 1196032);   // 4096*128*4   = 2097152

    cfconv_prep<<<(B_ * NA * NF + 255) / 256, 256, 0, stream>>>(
        x, W_in2f, W_f1, W_f2, W_out, xh, Win2fT, Wf1T, Wf2T, WoutT);
    cfconv_ygemm<<<(B_ * NA) / 64, 256, 0, stream>>>(xh, Win2fT, y);
    cfconv_main<<<B_ * NA, 256, 0, stream>>>(
        r_ij, neighbors, pmask, f_ij, b_f1, b_f2, b_out, Wf1T, Wf2T, WoutT, y, out);
}
